// TransformerRelativeBlock_3384434229570
// MI455X (gfx1250) — matrix operand order repaired and checked by reading, not compiled, not run on hardware
//
#include <hip/hip_runtime.h>
#include <cmath>
#include <cstdint>

// ---------------------------------------------------------------------------
// CDNA5 (gfx1250) fused transformer block with relative attention.
// Heavy math: rel-score term is algebraically refactored so the 512MB r tensor
// is streamed exactly once, via async global->LDS double buffering, through
// bf16 WMMAs (no rr materialization).
// ---------------------------------------------------------------------------

typedef __attribute__((ext_vector_type(16))) __bf16 v16bf;
typedef __attribute__((ext_vector_type(8)))  float  v8f;

#define FLAG_BIAS 1
#define FLAG_RES  2
#define FLAG_GELU 4

static __device__ __forceinline__ __bf16 f2bf(float x) { return (__bf16)x; }

// A fragment: 16(M) x 32(K) bf16 from row-major f32, row stride lda.
// Layout: lanes 0-15 row M=lane, K=0..15; lanes 16-31 same rows,
// K=16..31; VGPR v holds 2 consecutive K.
static __device__ __forceinline__ v16bf load_a_frag(const float* __restrict__ A, int lda) {
  int lane = threadIdx.x & 31;
  int hl = lane >> 4, m = lane & 15;
  const float* row = A + (long long)m * lda;
  v16bf a;
#pragma unroll
  for (int v = 0; v < 8; ++v) {
    int kk = 2 * v + 16 * hl;
    float2 t = *(const float2*)(row + kk);
    a[2 * v] = f2bf(t.x);
    a[2 * v + 1] = f2bf(t.y);
  }
  return a;
}

// B fragment: 32(K) x 16(N) bf16, B[k,n] = B[k*ldb + n] (row-major K x N).
// Layout used here (contiguous k): lanes 0-15 col N=lane hold K=0..15; lanes 16-31 hold K=16..31.
static __device__ __forceinline__ v16bf load_b_frag(const float* __restrict__ B, int ldb) {
  int lane = threadIdx.x & 31;
  int hl = lane >> 4, n = lane & 15;
  v16bf b;
#pragma unroll
  for (int v = 0; v < 8; ++v) {
    int kk = 2 * v + 16 * hl;
    b[2 * v]     = f2bf(B[(long long)kk * ldb + n]);
    b[2 * v + 1] = f2bf(B[(long long)(kk + 1) * ldb + n]);
  }
  return b;
}

// B fragment, transposed source: B[k,n] = B[n*ldb + k].
static __device__ __forceinline__ v16bf load_bT_frag(const float* __restrict__ B, int ldb) {
  int lane = threadIdx.x & 31;
  int hl = lane >> 4, n = lane & 15;
  const float* col = B + (long long)n * ldb;
  v16bf b;
#pragma unroll
  for (int v = 0; v < 8; ++v) {
    int kk = 2 * v + 16 * hl;
    float2 t = *(const float2*)(col + kk);
    b[2 * v]     = f2bf(t.x);
    b[2 * v + 1] = f2bf(t.y);
  }
  return b;
}

// ---------------------------------------------------------------------------
// Generic batched WMMA GEMM: C = epilogue(A @ B (+bias) (gelu) (+Res))
// Batch z decomposed as (bb, hh) with per-operand strides (handles per-head
// views into (B,T,D) tensors). One 16x16 tile per wave, 8 waves/block.
// M,N multiples of 16; K multiple of 32.
// ---------------------------------------------------------------------------
__global__ __launch_bounds__(256)
void wmma_gemm_kernel(const float* __restrict__ A, long long sAb, long long sAh, int lda,
                      const float* __restrict__ B, long long sBb, long long sBh, int ldb, int transB,
                      const float* __restrict__ bias,
                      const float* __restrict__ Res, long long sRb, long long sRh, int ldr,
                      float* __restrict__ C, long long sCb, long long sCh, int ldc,
                      int M, int N, int K, int Hdiv, int flags)
{
  int z = blockIdx.z;
  int bb = z / Hdiv, hh = z % Hdiv;
  A += bb * sAb + hh * sAh;
  B += bb * sBb + hh * sBh;
  C += bb * sCb + hh * sCh;
  if (Res) Res += bb * sRb + hh * sRh;

  int wave = threadIdx.x >> 5;
  int mt = blockIdx.x;
  int nt = blockIdx.y * 8 + wave;
  if (mt * 16 >= M || nt * 16 >= N) return;   // wave-uniform: EXEC all-1 at WMMA

  v8f acc = {0.f, 0.f, 0.f, 0.f, 0.f, 0.f, 0.f, 0.f};
  for (int kb = 0; kb < K; kb += 32) {
    v16bf a = load_a_frag(A + (long long)mt * 16 * lda + kb, lda);
    v16bf b = transB ? load_bT_frag(B + (long long)nt * 16 * ldb + kb, ldb)
                     : load_b_frag(B + (long long)kb * ldb + nt * 16, ldb);
    acc = __builtin_amdgcn_wmma_f32_16x16x32_bf16(false, a, false, b, (short)0, acc,
                                                  false, false);
  }

  int lane = threadIdx.x & 31;
  int hl = lane >> 4;
  int nn = nt * 16 + (lane & 15);
  float bval = (flags & FLAG_BIAS) ? bias[nn] : 0.f;
#pragma unroll
  for (int g = 0; g < 8; ++g) {
    int m = mt * 16 + g + 8 * hl;
    float vv = acc[g] + bval;
    if (flags & FLAG_GELU) vv = 0.5f * vv * (1.0f + erff(vv * 0.70710678118654752f));
    if (flags & FLAG_RES)  vv += Res[(long long)m * ldr + nn];
    C[(long long)m * ldc + nn] = vv;
  }
}

// ---------------------------------------------------------------------------
// LayerNorm: one block per row, blockDim == D (256).
// ---------------------------------------------------------------------------
__global__ __launch_bounds__(256)
void ln_kernel(const float* __restrict__ X, const float* __restrict__ g,
               const float* __restrict__ b, float* __restrict__ Y, int D)
{
  long long row = blockIdx.x;
  int t = threadIdx.x;
  float v = X[row * D + t];
  __shared__ float red[256];
  red[t] = v;
  __syncthreads();
  for (int s = 128; s > 0; s >>= 1) { if (t < s) red[t] += red[t + s]; __syncthreads(); }
  float mean = red[0] / (float)D;
  __syncthreads();
  float d = v - mean;
  red[t] = d * d;
  __syncthreads();
  for (int s = 128; s > 0; s >>= 1) { if (t < s) red[t] += red[t + s]; __syncthreads(); }
  float inv = rsqrtf(red[0] / (float)D + 1e-5f);
  Y[row * D + t] = d * inv * g[t] + b[t];
}

// ---------------------------------------------------------------------------
// Precompute per-(b,i,h):  u[d] = sum_{d'} Wr[d, h*dk+d'] * (q+rb)[b,h,i,d']
//                          c    = sum_{d'} br[h*dk+d']    * (q+rb)[b,h,i,d']
// This algebraic refactor kills the 68.7 GFLOP r@Wr GEMM and ~1GB of rr traffic.
// blockDim == D (256).
// ---------------------------------------------------------------------------
__global__ __launch_bounds__(256)
void rel_proj_kernel(const float* __restrict__ q, const float* __restrict__ rbias,
                     const float* __restrict__ Wr, const float* __restrict__ br,
                     float* __restrict__ U, float* __restrict__ Ca,
                     int D, int Hn, int dk)
{
  long long bi = blockIdx.x;              // b*T + i
  int t = threadIdx.x;                    // 0..D-1   (t == h*dk + d')
  __shared__ float tsh[256];
  tsh[t] = q[bi * D + t] + rbias[t];      // r_bias flat (1,H,1,dk) == D elems
  __syncthreads();
  const float* wrow = Wr + (long long)t * D;
  for (int h = 0; h < Hn; ++h) {
    float s = 0.f;
    for (int e = 0; e < dk; ++e) s += wrow[h * dk + e] * tsh[h * dk + e];
    U[(bi * Hn + h) * D + t] = s;
  }
  if (t < Hn) {
    float s = 0.f;
    for (int e = 0; e < dk; ++e) s += br[t * dk + e] * tsh[t * dk + e];
    Ca[bi * Hn + t] = s;
  }
}

// ---------------------------------------------------------------------------
// Streaming rel-score kernel: reads all of r exactly once via ASYNC global->LDS
// double buffering (ASYNCcnt-tracked), computes rel scores with WMMA from LDS.
//
// Block = (b,i); 8 waves. Each wave owns 2x 4KB LDS buffers (64KB/block) and
// walks its 16-j tiles in 16x64-float chunks:
//   issue chunk s+1 (8x global_load_async_to_lds_b128) -> s_wait_asynccnt 8
//   -> 2 WMMAs on chunk s from LDS. Buffer reuse is safe: chunk s-1's DS reads
//   are consumed (DScnt-waited) by its WMMA before chunk s+1's writes issue.
// Epilogue RMWs scores: S = (qk + rel + c) * scale, mask applied.
// Assumes D == 256, T multiple of 128.
// ---------------------------------------------------------------------------
__global__ __launch_bounds__(256)
void rel_scores_kernel(const float* __restrict__ r, const float* __restrict__ U,
                       const float* __restrict__ Ca, const int* __restrict__ mask,
                       float* __restrict__ S, int Bn, int Hn, int T, int D, float scale)
{
  __shared__ float smem[16384];                 // 8 waves x 2 bufs x 16x64 f32 = 64KB
  long long bi = blockIdx.x;
  int b = (int)(bi / T), i = (int)(bi % T);
  int wave = threadIdx.x >> 5;
  int lane = threadIdx.x & 31;
  int hl = lane >> 4, n = lane & 15;

  const float* rbase = r + bi * T * D;          // r[b,i,j,d] = rbase[j*D + d]
  const float* ub = U + bi * Hn * D;
  unsigned long long rb64 = (unsigned long long)rbase;

  // Hoist all 8 B fragments (K = 256 = 8*32); cols >= Hn zero-padded.
  v16bf bfr[8];
#pragma unroll
  for (int kb8 = 0; kb8 < 8; ++kb8) {
#pragma unroll
    for (int v = 0; v < 8; ++v) {
      int kk = kb8 * 32 + 2 * v + 16 * hl;
      float b0 = 0.f, b1 = 0.f;
      if (n < Hn) {
        b0 = ub[(long long)n * D + kk];
        b1 = ub[(long long)n * D + kk + 1];
      }
      bfr[kb8][2 * v]     = f2bf(b0);
      bfr[kb8][2 * v + 1] = f2bf(b1);
    }
  }

  float* wbuf = &smem[wave * 2048];             // this wave's 2 buffers
  // Low 32 bits of a generic LDS pointer == wave-relative LDS byte address.
  unsigned int lds0 = (unsigned int)(uintptr_t)wbuf;

  const int ntiles  = T / 16;                   // 32
  const int nchunks = (ntiles / 8) * 4;         // 16 chunks of 16 rows x 64 cols

  unsigned int lcol   = (unsigned int)((lane & 15) * 16);   // byte offset in row
  unsigned int rowsub = (unsigned int)(lane >> 4);          // 0/1: row within pair
  unsigned int rowpitch = (unsigned int)(D * 4);            // global row pitch

  auto issue = [&](int s) {
    int jt = wave + (s >> 2) * 8;
    int kb = s & 3;
    unsigned int gbase = (unsigned int)(((jt * 16) * D + kb * 64) * 4)
                         + rowsub * rowpitch + lcol;
    unsigned int lbase = lds0 + ((unsigned int)(s & 1) << 12)   // buffer select (4KB)
                         + rowsub * 256u + lcol;
#pragma unroll
    for (int p = 0; p < 8; ++p) {
      unsigned int g = gbase + (unsigned int)(2 * p) * rowpitch;
      unsigned int l = lbase + (unsigned int)(2 * p) * 256u;
      asm volatile("global_load_async_to_lds_b128 %0, %1, %2"
                   :: "v"(l), "v"(g), "s"(rb64) : "memory");
    }
  };

  issue(0);
  v8f acc = {0.f, 0.f, 0.f, 0.f, 0.f, 0.f, 0.f, 0.f};
  for (int s = 0; s < nchunks; ++s) {
    if (s + 1 < nchunks) {
      issue(s + 1);
      asm volatile("s_wait_asynccnt 0x8" ::: "memory");   // chunk s landed in LDS
    } else {
      asm volatile("s_wait_asynccnt 0x0" ::: "memory");
    }
    const float* chunk = wbuf + ((s & 1) ? 1024 : 0);
    int kb = s & 3;
    v16bf a0 = load_a_frag(chunk, 64);
    acc = __builtin_amdgcn_wmma_f32_16x16x32_bf16(false, a0, false, bfr[2 * kb],
                                                  (short)0, acc, false, false);
    v16bf a1 = load_a_frag(chunk + 32, 64);
    acc = __builtin_amdgcn_wmma_f32_16x16x32_bf16(false, a1, false, bfr[2 * kb + 1],
                                                  (short)0, acc, false, false);
    if (kb == 3) {                               // tile done -> epilogue
      int jt = wave + (s >> 2) * 8;
      if (n < Hn) {
        float cc = Ca[bi * Hn + n];
        float* srow = S + ((long long)(b * Hn + n) * T + i) * T;
        const int* mrow = mask + ((long long)b * T + i) * T;
#pragma unroll
        for (int g = 0; g < 8; ++g) {
          int j = jt * 16 + g + 8 * hl;          // C layout: row g + 8*half
          float sv = (srow[j] + acc[g] + cc) * scale;
          if (mrow[j] == 0) sv = -1e9f;
          srow[j] = sv;
        }
      }
      acc = (v8f){0.f, 0.f, 0.f, 0.f, 0.f, 0.f, 0.f, 0.f};
    }
  }
}

// ---------------------------------------------------------------------------
// Row softmax over last dim (in place). One block per row.
// ---------------------------------------------------------------------------
__global__ __launch_bounds__(256)
void softmax_kernel(float* __restrict__ S, int T)
{
  long long row = blockIdx.x;
  float* s = S + row * T;
  int t = threadIdx.x;
  __shared__ float red[256];
  float lmax = -1e30f;
  for (int j = t; j < T; j += 256) lmax = fmaxf(lmax, s[j]);
  red[t] = lmax;
  __syncthreads();
  for (int st = 128; st > 0; st >>= 1) { if (t < st) red[t] = fmaxf(red[t], red[t + st]); __syncthreads(); }
  float rowmax = red[0];
  __syncthreads();
  float lsum = 0.f;
  for (int j = t; j < T; j += 256) { float e = expf(s[j] - rowmax); s[j] = e; lsum += e; }
  red[t] = lsum;
  __syncthreads();
  for (int st = 128; st > 0; st >>= 1) { if (t < st) red[t] += red[t + st]; __syncthreads(); }
  float inv = 1.0f / red[0];
  for (int j = t; j < T; j += 256) s[j] *= inv;
}

// ---------------------------------------------------------------------------
static void launch_gemm(hipStream_t stream,
                        const float* A, long long sAb, long long sAh, int lda,
                        const float* B, long long sBb, long long sBh, int ldb, int transB,
                        const float* bias,
                        const float* Res, long long sRb, long long sRh, int ldr,
                        float* C, long long sCb, long long sCh, int ldc,
                        int M, int N, int K, int Hdiv, int nbatch, int flags)
{
  dim3 grid(M / 16, (N / 16 + 7) / 8, nbatch);
  wmma_gemm_kernel<<<grid, 256, 0, stream>>>(A, sAb, sAh, lda,
                                             B, sBb, sBh, ldb, transB,
                                             bias, Res, sRb, sRh, ldr,
                                             C, sCb, sCh, ldc,
                                             M, N, K, Hdiv, flags);
}

extern "C" void kernel_launch(void* const* d_in, const int* in_sizes, int n_in,
                              void* d_out, int out_size, void* d_ws, size_t ws_size,
                              hipStream_t stream)
{
  const float* x    = (const float*)d_in[0];
  const float* r    = (const float*)d_in[1];
  const int*   mask = (const int*)d_in[2];
  const float* Wq = (const float*)d_in[5];
  const float* bq = (const float*)d_in[6];
  const float* Wk = (const float*)d_in[7];
  const float* bk = (const float*)d_in[8];
  const float* Wv = (const float*)d_in[9];
  const float* bv = (const float*)d_in[10];
  const float* Wr = (const float*)d_in[11];
  const float* br = (const float*)d_in[12];
  const float* rb = (const float*)d_in[13];
  const float* Wo = (const float*)d_in[14];
  const float* bo = (const float*)d_in[15];
  const float* g1 = (const float*)d_in[16];
  const float* b1 = (const float*)d_in[17];
  const float* g2 = (const float*)d_in[18];
  const float* b2 = (const float*)d_in[19];
  const float* W1 = (const float*)d_in[20];
  const float* fb1 = (const float*)d_in[21];
  const float* W2 = (const float*)d_in[22];
  const float* fb2 = (const float*)d_in[23];
  float* out = (float*)d_out;

  const int D = in_sizes[6];                       // bq: (D,)
  const long long xsz = in_sizes[0];               // B*T*D
  const long long rsz = in_sizes[1];               // B*T*T*D
  const int T  = (int)(rsz / xsz);
  const int Bn = (int)(xsz / ((long long)T * D));
  const int H  = 4;                                // r_bias (1,H,1,dk); flat size only gives H*dk
  const int dk = D / H;
  const int BT = Bn * T;
  const int D4 = 4 * D;
  const float scale = 1.0f / sqrtf((float)dk);

  // Workspace carve-up (f32): ~24 MB total for the default shapes.
  float* ws  = (float*)d_ws;
  float* h1  = ws; ws += (long long)BT * D;
  float* q   = ws; ws += (long long)BT * D;
  float* k   = ws; ws += (long long)BT * D;
  float* v   = ws; ws += (long long)BT * D;
  float* U   = ws; ws += (long long)BT * H * D;
  float* Ca  = ws; ws += (long long)BT * H;
  float* S   = ws; ws += (long long)Bn * H * T * T;
  float* ctx = ws; ws += (long long)BT * D;
  float* x1  = ws; ws += (long long)BT * D;
  float* h2  = ws; ws += (long long)BT * D;
  float* f1  = ws; /* BT * D4 */

  // 1) h1 = LN(x; g1,b1)
  ln_kernel<<<BT, D, 0, stream>>>(x, g1, b1, h1, D);

  // 2) q,k,v = h1 @ W{q,k,v} + b  (stored (B,T,D); head h at cols h*dk..)
  launch_gemm(stream, h1, 0, 0, D, Wq, 0, 0, D, 0, bq, nullptr, 0, 0, 0,
              q, 0, 0, D, BT, D, D, 1, 1, FLAG_BIAS);
  launch_gemm(stream, h1, 0, 0, D, Wk, 0, 0, D, 0, bk, nullptr, 0, 0, 0,
              k, 0, 0, D, BT, D, D, 1, 1, FLAG_BIAS);
  launch_gemm(stream, h1, 0, 0, D, Wv, 0, 0, D, 0, bv, nullptr, 0, 0, 0,
              v, 0, 0, D, BT, D, D, 1, 1, FLAG_BIAS);

  // 3) u,c per (b,i,h)  — algebraic refactor of the rel term
  rel_proj_kernel<<<BT, D, 0, stream>>>(q, rb, Wr, br, U, Ca, D, H, dk);

  // 4) base scores S[b,h] = q_bh @ k_bh^T  (unscaled)
  launch_gemm(stream, q, (long long)T * D, dk, D,
              k, (long long)T * D, dk, D, 1,
              nullptr, nullptr, 0, 0, 0,
              S, (long long)H * T * T, (long long)T * T, T,
              T, T, dk, H, Bn * H, 0);

  // 5) stream r once (async->LDS): S = (S + r·u + c) * scale, with mask
  rel_scores_kernel<<<BT, 256, 0, stream>>>(r, U, Ca, mask, S, Bn, H, T, D, scale);

  // 6) softmax over j
  softmax_kernel<<<Bn * H * T, 256, 0, stream>>>(S, T);

  // 7) ctx[b,h] = P_bh @ v_bh   (written into (B,T,D) head slots)
  launch_gemm(stream, S, (long long)H * T * T, (long long)T * T, T,
              v, (long long)T * D, dk, D, 0,
              nullptr, nullptr, 0, 0, 0,
              ctx, (long long)T * D, dk, D,
              T, dk, T, H, Bn * H, 0);

  // 8) x1 = x + ctx @ Wo + bo
  launch_gemm(stream, ctx, 0, 0, D, Wo, 0, 0, D, 0, bo, x, 0, 0, D,
              x1, 0, 0, D, BT, D, D, 1, 1, FLAG_BIAS | FLAG_RES);

  // 9) h2 = LN(x1; g2,b2)
  ln_kernel<<<BT, D, 0, stream>>>(x1, g2, b2, h2, D);

  // 10) f1 = gelu(h2 @ W1 + fb1)
  launch_gemm(stream, h2, 0, 0, D, W1, 0, 0, D4, 0, fb1, nullptr, 0, 0, 0,
              f1, 0, 0, D4, BT, D4, D, 1, 1, FLAG_BIAS | FLAG_GELU);

  // 11) out = x1 + f1 @ W2 + fb2
  launch_gemm(stream, f1, 0, 0, D4, W2, 0, 0, D, 0, fb2, x1, 0, 0, D,
              out, 0, 0, D, BT, D, D4, 1, 1, FLAG_BIAS | FLAG_RES);
}
